// GTS_Forecasting_Module_50783693308173
// MI455X (gfx1250) — matrix operand order repaired and checked by reading, not compiled, not run on hardware
//
#include <hip/hip_runtime.h>
#include <hip/hip_bf16.h>

// ---------------------------------------------------------------------------
// GTS forecasting (DCGRU enc-dec) for MI455X / gfx1250.
// Heavy op: adj(5000x5000) @ X(5000x33), 96 times.  adj converted once to
// bf16 (51 MB -> L2 resident), padded to 5056.  Diffusion products run on
// v_wmma_f32_16x16x32_bf16 with f32 accumulation.  Each wave register-blocks
// 4 M-tiles x 3 N-tiles (12 accumulators) so every B tile load feeds 4 WMMAs,
// cutting L2 feature traffic 4x (adj read exactly once per product).
// K split 8 ways into deterministic partial buffers (no atomics).
// ---------------------------------------------------------------------------

#define N_NODES 5000
#define NP      5056      // padded nodes: 158*32 (mult of 32 for K, 64 for M-groups)
#define U_RNN   32
#define FIN     33        // IN + U
#define FPAD    48        // 3 column-tiles of 16
#define DINF    99        // (K+1)*(IN+U)
#define T_ENC   12
#define T_DEC   12
#define NSPLIT  8
#define KCHUNK  640       // 20*32 ; last chunk = 5056-7*640 = 576 = 18*32
#define MGROUPS (NP / 64) // 79 groups of 4 row-tiles

typedef __attribute__((ext_vector_type(16))) __bf16 v16bf;
typedef __attribute__((ext_vector_type(8)))  float  v8f;

__device__ __forceinline__ unsigned short f32_to_bf16_bits(float f) {
    unsigned int u = __float_as_uint(f);
    u += 0x7FFFu + ((u >> 16) & 1u);          // round-to-nearest-even
    return (unsigned short)(u >> 16);
}

// ---------------- one-time: adj f32 [N][N] -> bf16 [NP][NP], zero padded ----
__global__ void k_convert_adj(const float* __restrict__ adj,
                              unsigned short* __restrict__ adjb) {
    long idx = (long)blockIdx.x * blockDim.x + threadIdx.x;
    const long total = (long)NP * NP;
    if (idx >= total) return;
    int i = (int)(idx / NP), j = (int)(idx % NP);
    float v = (i < N_NODES && j < N_NODES) ? adj[(long)i * N_NODES + j] : 0.0f;
    adjb[idx] = f32_to_bf16_bits(v);
}

__global__ void k_zero(float* __restrict__ p, int n) {
    int i = blockIdx.x * blockDim.x + threadIdx.x;
    if (i < n) p[i] = 0.0f;
}

// ------------- build xh = [x ; h] in feature-major f32 + bf16 ---------------
// xsrc indexed xsrc[n*xstride]  (encoder: inputs+t, stride T_ENC; decoder: x_sc, stride 1)
__global__ void k_build_xh(const float* __restrict__ xsrc, int xstride,
                           const float* __restrict__ h,
                           float* __restrict__ xh_f,
                           unsigned short* __restrict__ xh_b) {
    int n = blockIdx.x * blockDim.x + threadIdx.x;
    if (n >= NP) return;
    float x = (n < N_NODES) ? xsrc[(size_t)n * xstride] : 0.0f;
    xh_f[n] = x;
    xh_b[n] = f32_to_bf16_bits(x);
    #pragma unroll 4
    for (int i = 0; i < U_RNN; ++i) {
        float v = h[(size_t)i * NP + n];
        xh_f[(size_t)(1 + i) * NP + n] = v;
        xh_b[(size_t)(1 + i) * NP + n] = f32_to_bf16_bits(v);
    }
    for (int f = FIN; f < FPAD; ++f) {           // pad tiles 33..47 with zeros
        xh_f[(size_t)f * NP + n] = 0.0f;
        xh_b[(size_t)f * NP + n] = 0;
    }
}

// ---------------- Y = adj @ X  via WMMA bf16 -------------------------------
// A: adjb [NP][NP] bf16 row-major.  B: feature-major X [48][NP] bf16.
// Grid: (MGROUPS, NSPLIT); one wave per block.  Each wave: 4 M-tiles x 3 N-tiles.
// Output: partials[split][48][NP] f32 (feature-major, i.e. transposed store).
__global__ void __launch_bounds__(32)
k_adjmm(const unsigned short* __restrict__ A,
        const unsigned short* __restrict__ B,
        float* __restrict__ partials) {
    const int split = blockIdx.y;                // K split
    const int lane  = threadIdx.x & 31;
    const int g     = lane >> 4;                 // half-wave group
    const int lr    = lane & 15;
    const int m0    = blockIdx.x * 64;           // 4 row-tiles per wave

    int k0 = split * KCHUNK;
    int k1 = k0 + KCHUNK; if (k1 > NP) k1 = NP;

    v8f z = {0.f, 0.f, 0.f, 0.f, 0.f, 0.f, 0.f, 0.f};
    v8f acc[12];                                 // [mt][f]
    #pragma unroll
    for (int i = 0; i < 12; ++i) acc[i] = z;

    const unsigned short* arow[4];
    #pragma unroll
    for (int mt = 0; mt < 4; ++mt)
        arow[mt] = A + (size_t)(m0 + mt * 16 + lr) * NP;

    const unsigned short* brow[3];
    #pragma unroll
    for (int f = 0; f < 3; ++f)
        brow[f] = B + (size_t)(f * 16 + lr) * NP;

    union UAB { v16bf v; uint4 q[2]; };

    for (int k = k0; k < k1; k += 32) {
        // B tiles 32x16: lane = feature column, K values k+g*8..+7 and k+16+g*8..+7
        UAB b[3];
        #pragma unroll
        for (int f = 0; f < 3; ++f) {
            b[f].q[0] = *reinterpret_cast<const uint4*>(brow[f] + k + g * 8);
            b[f].q[1] = *reinterpret_cast<const uint4*>(brow[f] + k + 16 + g * 8);
        }
        // A tiles 16x32, reused across the 3 N-tiles
        #pragma unroll
        for (int mt = 0; mt < 4; ++mt) {
            UAB a;
            a.q[0] = *reinterpret_cast<const uint4*>(arow[mt] + k + g * 8);
            a.q[1] = *reinterpret_cast<const uint4*>(arow[mt] + k + 16 + g * 8);
            __builtin_prefetch(arow[mt] + k + 1024, 0, 0);  // stream adj (global_prefetch_b8)
            #pragma unroll
            for (int f = 0; f < 3; ++f) {
                acc[mt * 3 + f] = __builtin_amdgcn_wmma_f32_16x16x32_bf16(
                    false, a.v, false, b[f].v, (short)0, acc[mt * 3 + f], false, false);
            }
        }
    }

    // Transposed (feature-major) store: lane's 8 accum values have consecutive M.
    union UC { v8f v; float4 q[2]; };
    float* pbase = partials + (size_t)split * FPAD * NP;
    #pragma unroll
    for (int mt = 0; mt < 4; ++mt) {
        #pragma unroll
        for (int f = 0; f < 3; ++f) {
            UC c; c.v = acc[mt * 3 + f];
            float* p = pbase + (size_t)(f * 16 + lr) * NP + m0 + mt * 16 + g * 8;
            reinterpret_cast<float4*>(p)[0] = c.q[0];
            reinterpret_cast<float4*>(p)[1] = c.q[1];
        }
    }
}

// -------- reduce K-split partials; emit f32 (for GEMMs) + bf16 (next hop) ---
__global__ void k_reduce(const float* __restrict__ partials,
                         float* __restrict__ out_f,
                         unsigned short* __restrict__ out_b) {
    int idx = blockIdx.x * blockDim.x + threadIdx.x;
    if (idx >= FPAD * NP) return;
    float s = 0.0f;
    #pragma unroll
    for (int w = 0; w < NSPLIT; ++w) s += partials[(size_t)w * FPAD * NP + idx];
    out_f[idx] = s;
    out_b[idx] = f32_to_bf16_bits(s);
}

// -------- gate: ru = sigmoid(concat(xh,x1,x2) @ Wg + bg); emit u, r*h -------
__global__ void __launch_bounds__(256)
k_gate(const float* __restrict__ xh, const float* __restrict__ x1,
       const float* __restrict__ x2,
       const float* __restrict__ Wg, const float* __restrict__ bg,
       const float* __restrict__ h,
       float* __restrict__ u_fm,
       float* __restrict__ xh_f, unsigned short* __restrict__ xh_b) {
    __shared__ float sW[DINF * 2 * U_RNN];       // 99x64 = 25.3 KB (LDS is 320KB/WGP)
    __shared__ float sb[2 * U_RNN];
    for (int i = threadIdx.x; i < DINF * 2 * U_RNN; i += blockDim.x) sW[i] = Wg[i];
    for (int i = threadIdx.x; i < 2 * U_RNN; i += blockDim.x) sb[i] = bg[i];
    __syncthreads();
    int n = blockIdx.x * blockDim.x + threadIdx.x;
    if (n >= N_NODES) return;

    float acc[2 * U_RNN];
    #pragma unroll
    for (int o = 0; o < 2 * U_RNN; ++o) acc[o] = sb[o];

    const float* bases[3] = { xh, x1, x2 };
    for (int s3 = 0; s3 < 3; ++s3) {
        const float* src  = bases[s3];
        const float* wblk = sW + s3 * FIN * 2 * U_RNN;
        for (int f = 0; f < FIN; ++f) {
            float v = src[(size_t)f * NP + n];
            const float* wrow = wblk + f * 2 * U_RNN;
            #pragma unroll
            for (int o = 0; o < 2 * U_RNN; ++o) acc[o] = fmaf(v, wrow[o], acc[o]);
        }
    }
    #pragma unroll 4
    for (int i = 0; i < U_RNN; ++i) {
        float r  = 1.0f / (1.0f + __expf(-acc[i]));
        float u  = 1.0f / (1.0f + __expf(-acc[U_RNN + i]));
        float rh = r * h[(size_t)i * NP + n];
        u_fm[(size_t)i * NP + n] = u;
        xh_f[(size_t)(1 + i) * NP + n] = rh;         // row 0 (x) stays in place
        xh_b[(size_t)(1 + i) * NP + n] = f32_to_bf16_bits(rh);
    }
}

// -------- candidate + GRU update: h = u*h + (1-u)*tanh(concat @ Wc + bc) ----
__global__ void __launch_bounds__(256)
k_cand(const float* __restrict__ xh, const float* __restrict__ x1,
       const float* __restrict__ x2,
       const float* __restrict__ Wc, const float* __restrict__ bc,
       const float* __restrict__ u_fm,
       float* __restrict__ h) {
    __shared__ float sW[DINF * U_RNN];           // 99x32
    __shared__ float sb[U_RNN];
    for (int i = threadIdx.x; i < DINF * U_RNN; i += blockDim.x) sW[i] = Wc[i];
    for (int i = threadIdx.x; i < U_RNN; i += blockDim.x) sb[i] = bc[i];
    __syncthreads();
    int n = blockIdx.x * blockDim.x + threadIdx.x;
    if (n >= N_NODES) return;

    float acc[U_RNN];
    #pragma unroll
    for (int o = 0; o < U_RNN; ++o) acc[o] = sb[o];

    const float* bases[3] = { xh, x1, x2 };
    for (int s3 = 0; s3 < 3; ++s3) {
        const float* src  = bases[s3];
        const float* wblk = sW + s3 * FIN * U_RNN;
        for (int f = 0; f < FIN; ++f) {
            float v = src[(size_t)f * NP + n];
            const float* wrow = wblk + f * U_RNN;
            #pragma unroll
            for (int o = 0; o < U_RNN; ++o) acc[o] = fmaf(v, wrow[o], acc[o]);
        }
    }
    #pragma unroll 4
    for (int i = 0; i < U_RNN; ++i) {
        float c  = tanhf(acc[i]);
        float u  = u_fm[(size_t)i * NP + n];
        float hv = h[(size_t)i * NP + n];
        h[(size_t)i * NP + n] = u * hv + (1.0f - u) * c;
    }
}

// -------- decoder projection: out = h @ Wp + bp ; also feeds next x ---------
__global__ void k_project(const float* __restrict__ h,
                          const float* __restrict__ Wp, const float* __restrict__ bp,
                          float* __restrict__ d_out, float* __restrict__ x_sc, int t) {
    int n = blockIdx.x * blockDim.x + threadIdx.x;
    if (n >= N_NODES) return;
    float acc = bp[0];
    #pragma unroll
    for (int i = 0; i < U_RNN; ++i)
        acc = fmaf(h[(size_t)i * NP + n], Wp[i], acc);
    d_out[(size_t)n * T_DEC + t] = acc;
    x_sc[n] = acc;
}

// ---------------------------------------------------------------------------
extern "C" void kernel_launch(void* const* d_in, const int* in_sizes, int n_in,
                              void* d_out, int out_size, void* d_ws, size_t ws_size,
                              hipStream_t stream) {
    (void)in_sizes; (void)n_in; (void)out_size; (void)ws_size;

    const float* inputs = (const float*)d_in[0];
    // d_in[1] = targets (unused in eval mode)
    const float* adj    = (const float*)d_in[2];
    const float* enc_Wg = (const float*)d_in[3];
    const float* enc_bg = (const float*)d_in[4];
    const float* enc_Wc = (const float*)d_in[5];
    const float* enc_bc = (const float*)d_in[6];
    const float* dec_Wg = (const float*)d_in[7];
    const float* dec_bg = (const float*)d_in[8];
    const float* dec_Wc = (const float*)d_in[9];
    const float* dec_bc = (const float*)d_in[10];
    const float* dec_Wp = (const float*)d_in[11];
    const float* dec_bp = (const float*)d_in[12];
    float* out = (float*)d_out;

    // ---- carve workspace (256B aligned slabs) ----
    char* wsp = (char*)d_ws;
    auto alloc = [&](size_t bytes) -> char* {
        char* p = wsp;
        wsp += (bytes + 255) & ~(size_t)255;
        return p;
    };
    unsigned short* adjb = (unsigned short*)alloc((size_t)NP * NP * 2);   // 51 MB (L2 resident)
    unsigned short* xh_b = (unsigned short*)alloc((size_t)FPAD * NP * 2);
    unsigned short* x1_b = (unsigned short*)alloc((size_t)FPAD * NP * 2);
    unsigned short* x2_b = (unsigned short*)alloc((size_t)FPAD * NP * 2);
    float* xh_f = (float*)alloc((size_t)FPAD * NP * 4);
    float* x1_f = (float*)alloc((size_t)FPAD * NP * 4);
    float* x2_f = (float*)alloc((size_t)FPAD * NP * 4);
    float* part = (float*)alloc((size_t)NSPLIT * FPAD * NP * 4);
    float* h_fm = (float*)alloc((size_t)U_RNN * NP * 4);
    float* u_fm = (float*)alloc((size_t)U_RNN * NP * 4);
    float* x_sc = (float*)alloc((size_t)NP * 4);

    const int  RN   = FPAD * NP;
    const long ATOT = (long)NP * NP;
    const dim3 mmGrid(MGROUPS, NSPLIT);

    // one-time prep
    k_convert_adj<<<(unsigned)((ATOT + 255) / 256), 256, 0, stream>>>(adj, adjb);
    k_zero<<<(U_RNN * NP + 255) / 256, 256, 0, stream>>>(h_fm, U_RNN * NP);
    k_zero<<<(NP + 255) / 256, 256, 0, stream>>>(x_sc, NP);

    auto dcgru_step = [&](const float* xsrc, int xstride,
                          const float* Wg, const float* bg,
                          const float* Wc, const float* bc) {
        k_build_xh<<<(NP + 255) / 256, 256, 0, stream>>>(xsrc, xstride, h_fm, xh_f, xh_b);
        // gate diffusion: x1 = A@xh, x2 = A@x1
        k_adjmm<<<mmGrid, 32, 0, stream>>>(adjb, xh_b, part);
        k_reduce<<<(RN + 255) / 256, 256, 0, stream>>>(part, x1_f, x1_b);
        k_adjmm<<<mmGrid, 32, 0, stream>>>(adjb, x1_b, part);
        k_reduce<<<(RN + 255) / 256, 256, 0, stream>>>(part, x2_f, x2_b);
        k_gate<<<(N_NODES + 255) / 256, 256, 0, stream>>>(xh_f, x1_f, x2_f, Wg, bg,
                                                          h_fm, u_fm, xh_f, xh_b);
        // candidate diffusion on [x ; r*h]
        k_adjmm<<<mmGrid, 32, 0, stream>>>(adjb, xh_b, part);
        k_reduce<<<(RN + 255) / 256, 256, 0, stream>>>(part, x1_f, x1_b);
        k_adjmm<<<mmGrid, 32, 0, stream>>>(adjb, x1_b, part);
        k_reduce<<<(RN + 255) / 256, 256, 0, stream>>>(part, x2_f, x2_b);
        k_cand<<<(N_NODES + 255) / 256, 256, 0, stream>>>(xh_f, x1_f, x2_f, Wc, bc,
                                                          u_fm, h_fm);
    };

    // encoder
    for (int t = 0; t < T_ENC; ++t)
        dcgru_step(inputs + t, T_ENC, enc_Wg, enc_bg, enc_Wc, enc_bc);

    // decoder (autoregressive, feeds own projection)
    for (int t = 0; t < T_DEC; ++t) {
        dcgru_step(x_sc, 1, dec_Wg, dec_bg, dec_Wc, dec_bc);
        k_project<<<(N_NODES + 255) / 256, 256, 0, stream>>>(h_fm, dec_Wp, dec_bp,
                                                             out, x_sc, t);
    }
}
